// BestRqLossNetwork_57973468561599
// MI455X (gfx1250) — matrix operand order repaired and checked by reading, not compiled, not run on hardware
//
#include <hip/hip_runtime.h>
#include <hip/hip_bf16.h>
#include <math.h>

typedef unsigned short u16;
typedef unsigned int   u32;

typedef __attribute__((ext_vector_type(16))) __bf16 v16bf;
typedef __attribute__((ext_vector_type(8)))  float  v8f;

union BF16Frag { v16bf v; uint4 q[2]; };

#define SA_STRIDE 520   // 512 + 8 bf16 pad -> 1040B row stride, conflict-free b128 LDS reads

__device__ __forceinline__ u16 f2bf(float f) {
    union { float f; u32 u; } c; c.f = f;
    u32 u = c.u;
    u32 r = u + 0x7FFFu + ((u >> 16) & 1u);   // round-to-nearest-even
    return (u16)(r >> 16);
}

// ---------------------------------------------------------------------------
// Kernel 0a: context (f32, row-major 8192x512) -> bf16, same layout
// ---------------------------------------------------------------------------
__global__ void cvtA_kernel(const float* __restrict__ src, u16* __restrict__ dst) {
    int i = (blockIdx.x * 256 + threadIdx.x) * 4;     // n = 4,194,304, multiple of 4
    float4 x = *(const float4*)(src + i);
    ushort4 o;
    o.x = f2bf(x.x); o.y = f2bf(x.y); o.z = f2bf(x.z); o.w = f2bf(x.w);
    *(ushort4*)(dst + i) = o;
}

// ---------------------------------------------------------------------------
// Kernel 0b: W_enc (f32, [c=512][k=8192]) -> bf16 transposed:
//            Wt[(k)*512 + c], so each lane's WMMA B-fragment is two 16-byte
//            contiguous runs of K for one output column.
// ---------------------------------------------------------------------------
__global__ void cvtBT_kernel(const float* __restrict__ W, u16* __restrict__ Wt) {
    __shared__ float tile[32][33];
    int kt = blockIdx.x;                 // 0..255  (k tiles of 32)
    int ct = blockIdx.y;                 // 0..15   (c tiles of 32)
    int tx = threadIdx.x & 31;
    int ty = threadIdx.x >> 5;           // 0..7
    int k0 = kt * 32, c0 = ct * 32;
#pragma unroll
    for (int i = 0; i < 32; i += 8)
        tile[ty + i][tx] = W[(size_t)(c0 + ty + i) * 8192 + k0 + tx];   // tile[c_local][k_local]
    __syncthreads();
#pragma unroll
    for (int i = 0; i < 32; i += 8)
        Wt[(size_t)(k0 + ty + i) * 512 + c0 + tx] = f2bf(tile[tx][ty + i]);
}

// ---------------------------------------------------------------------------
// Kernel 1: targets.  Block = 16 rows x 16 threads/row.
//   f = feats_row(512) @ proj(512x16); normalize; argmin_k ||f - c_k||^2
// ---------------------------------------------------------------------------
__global__ void targets_kernel(const float* __restrict__ feats,
                               const int*   __restrict__ lens,
                               const float* __restrict__ proj,
                               const float* __restrict__ codebook,
                               int* __restrict__ targets) {
    __shared__ float sP[512 * 16];     // 32KB projection matrix
    __shared__ float sF[16][16];
    __shared__ float sRed[16][16];
    __shared__ int   sRedI[16][16];

    int tid = threadIdx.x;             // 256
    int r0  = blockIdx.x * 16;
    for (int i = tid; i < 512 * 16; i += 256) sP[i] = proj[i];
    __syncthreads();

    int row = tid >> 4, j = tid & 15;
    const float* fr = feats + (size_t)(r0 + row) * 512;
    float acc = 0.f;
    for (int c = 0; c < 512; ++c) acc += fr[c] * sP[c * 16 + j];
    sRed[row][j] = acc * acc;
    __syncthreads();
    if (j == 0) {
        float ss = 0.f;
        for (int q = 0; q < 16; ++q) ss += sRed[row][q];
        sRed[row][0] = 1.0f / fmaxf(sqrtf(ss), 1e-12f);
    }
    __syncthreads();
    sF[row][j] = acc * sRed[row][0];
    __syncthreads();

    float f[16];
#pragma unroll
    for (int i = 0; i < 16; ++i) f[i] = sF[row][i];

    float bestd = __builtin_inff();
    int   bestk = 0;
    for (int it = 0; it < 512; ++it) {
        int k = it * 16 + j;                       // lanes scan consecutive entries
        const float* cbp = codebook + (size_t)k * 16;
        float dot = 0.f, cc = 0.f;
#pragma unroll
        for (int i = 0; i < 16; ++i) { float cv = cbp[i]; dot += f[i] * cv; cc += cv * cv; }
        float d = cc - 2.0f * dot;                 // + |f|^2 (constant per row)
        if (d < bestd || (d == bestd && k < bestk)) { bestd = d; bestk = k; }
    }
    sRed[row][j] = bestd; sRedI[row][j] = bestk;
    __syncthreads();
    if (j == 0) {
        float bd = sRed[row][0]; int bk = sRedI[row][0];
        for (int q = 1; q < 16; ++q) {
            float d = sRed[row][q]; int k = sRedI[row][q];
            if (d < bd || (d == bd && k < bk)) { bd = d; bk = k; }
        }
        int r = r0 + row;
        int n = r >> 11, t = r & 2047;
        targets[r] = (t < lens[n]) ? bk : -100;
    }
}

// ---------------------------------------------------------------------------
// Kernel 2: fused bf16 WMMA GEMM (logits) + online-softmax cross entropy.
//   Block = 32 rows, 8 waves; 64 column-blocks of 128 cols (16 per wave).
//   Each B fragment feeds TWO WMMAs (rows 0-15 and 16-31): halves L2
//   B-traffic and the VMEM-issue : WMMA ratio vs a 16-row tile.
// ---------------------------------------------------------------------------
__global__ void gemmce_kernel(const u16* __restrict__ Abf,   // [8192][512] bf16
                              const u16* __restrict__ Wt,    // [8192 cols][512 k] bf16
                              const float* __restrict__ bias,
                              const int* __restrict__ targets,
                              float* __restrict__ partials)  // [256][2] = {sum_nll, n_valid}
{
    __shared__ u16   sA[32 * SA_STRIDE];   // 32.5KB padded A panel (32 rows)
    __shared__ float sTile[32][128];       // logits staging (16KB)
    __shared__ float sRed[32][8];
    __shared__ float sM[32], sS[32], sNewM[32], sTgt[32];
    __shared__ int   sTv[32];

    int tid = threadIdx.x;                 // 256 = 8 waves
    int wg  = blockIdx.x;                  // 256 workgroups
    int r0  = wg * 32;

    // A panel: rows r0..r0+31, each 512 bf16 contiguous => one 32KB burst
    {
        const uint4* src = (const uint4*)(Abf + (size_t)r0 * 512);   // 2048 x 16B
        for (int i = tid; i < 2048; i += 256) {
            int rowi = i >> 6;             // 64 uint4 per row
            int colq = i & 63;
            *(uint4*)(sA + rowi * SA_STRIDE + colq * 8) = src[i];
        }
    }
    if (tid < 32) {
        sM[tid] = -__builtin_inff(); sS[tid] = 0.f; sTgt[tid] = 0.f;
        sTv[tid] = targets[r0 + tid];
    }
    __syncthreads();

    int wave = tid >> 5;
    int lane = tid & 31;
    int hl   = lane >> 4;                  // lane half (ISA fragment layout)
    int l16  = lane & 15;
    const u16* apBase0 = sA + l16 * SA_STRIDE + hl * 8;          // rows 0-15
    const u16* apBase1 = sA + (16 + l16) * SA_STRIDE + hl * 8;   // rows 16-31
    int row = tid >> 3, j = tid & 7;       // epilogue roles: 32 rows x 8 threads

    for (int cbI = 0; cbI < 64; ++cbI) {
        int col = cbI * 128 + wave * 16 + l16;
        const u16* bp = Wt + (size_t)col * 512 + hl * 8;   // B: K 8h..8h+7 and 16+8h..16+8h+7
        if (cbI + 1 < 64)                                  // stream-ahead hint for next block
            __builtin_prefetch(bp + 128 * 512, 0, 0);      // -> global_prefetch_b8
        v8f acc0 = {0.f, 0.f, 0.f, 0.f, 0.f, 0.f, 0.f, 0.f};
        v8f acc1 = {0.f, 0.f, 0.f, 0.f, 0.f, 0.f, 0.f, 0.f};
#pragma unroll
        for (int i = 0; i < 16; ++i) {                 // k0 = 32*i
            BF16Frag b, a0, a1;
            b.q[0]  = *(const uint4*)(bp + 32 * i);
            b.q[1]  = *(const uint4*)(bp + 32 * i + 16);
            a0.q[0] = *(const uint4*)(apBase0 + 32 * i);
            a0.q[1] = *(const uint4*)(apBase0 + 32 * i + 16);
            a1.q[0] = *(const uint4*)(apBase1 + 32 * i);
            a1.q[1] = *(const uint4*)(apBase1 + 32 * i + 16);
            acc0 = __builtin_amdgcn_wmma_f32_16x16x32_bf16(
                false, a0.v, false, b.v, (short)0, acc0, false, false);
            acc1 = __builtin_amdgcn_wmma_f32_16x16x32_bf16(
                false, a1.v, false, b.v, (short)0, acc1, false, false);
        }
        float bv = bias[col];
#pragma unroll
        for (int i = 0; i < 8; ++i) {                  // D row = i + 8*hl (+16 for acc1)
            sTile[i + 8 * hl][wave * 16 + l16]      = acc0[i] + bv;
            sTile[16 + i + 8 * hl][wave * 16 + l16] = acc1[i] + bv;
        }
        __syncthreads();

        // --- online softmax update over this 128-column block ---
        float v[16];
        {
            float4 q0 = *(const float4*)&sTile[row][j * 16];
            float4 q1 = *(const float4*)&sTile[row][j * 16 + 4];
            float4 q2 = *(const float4*)&sTile[row][j * 16 + 8];
            float4 q3 = *(const float4*)&sTile[row][j * 16 + 12];
            v[0]=q0.x; v[1]=q0.y; v[2]=q0.z; v[3]=q0.w;
            v[4]=q1.x; v[5]=q1.y; v[6]=q1.z; v[7]=q1.w;
            v[8]=q2.x; v[9]=q2.y; v[10]=q2.z; v[11]=q2.w;
            v[12]=q3.x; v[13]=q3.y; v[14]=q3.z; v[15]=q3.w;
        }
        float lmax = v[0];
#pragma unroll
        for (int q = 1; q < 16; ++q) lmax = fmaxf(lmax, v[q]);
        sRed[row][j] = lmax;
        __syncthreads();
        if (j == 0) {
            float bm = sRed[row][0];
            for (int q = 1; q < 8; ++q) bm = fmaxf(bm, sRed[row][q]);
            sNewM[row] = fmaxf(sM[row], bm);
        }
        __syncthreads();
        float nm = sNewM[row];
        float ls = 0.f;
#pragma unroll
        for (int q = 0; q < 16; ++q) ls += __expf(v[q] - nm);
        sRed[row][j] = ls;
        int tcol = sTv[row] - cbI * 128;               // capture target logit
        if (tcol >= 0 && tcol < 128 && (tcol >> 4) == j) sTgt[row] = v[tcol & 15];
        __syncthreads();
        if (j == 0) {
            float om = sM[row];
            float ns = (om == -__builtin_inff()) ? 0.f : sS[row] * __expf(om - nm);
            for (int q = 0; q < 8; ++q) ns += sRed[row][q];
            sS[row] = ns; sM[row] = nm;
        }
        __syncthreads();
    }

    if (tid < 32) {
        float rl = 0.f, val = 0.f;
        if (sTv[tid] != -100) { rl = sM[tid] + __logf(sS[tid]) - sTgt[tid]; val = 1.f; }
        sM[tid] = rl; sS[tid] = val;       // reuse stats arrays for final reduce
    }
    __syncthreads();
    if (tid == 0) {
        float s = 0.f, c = 0.f;
        for (int q = 0; q < 32; ++q) { s += sM[q]; c += sS[q]; }
        partials[wg * 2]     = s;
        partials[wg * 2 + 1] = c;
    }
}

// ---------------------------------------------------------------------------
// Kernel 3: deterministic final reduction -> loss scalar
// ---------------------------------------------------------------------------
__global__ void finalize_kernel(const float* __restrict__ partials, float* __restrict__ out) {
    __shared__ float ss[256], sc[256];
    int tid = threadIdx.x;
    float s = 0.f, c = 0.f;
    for (int i = tid; i < 256; i += 256) { s += partials[2 * i]; c += partials[2 * i + 1]; }
    ss[tid] = s; sc[tid] = c;
    __syncthreads();
    for (int off = 128; off > 0; off >>= 1) {
        if (tid < off) { ss[tid] += ss[tid + off]; sc[tid] += sc[tid + off]; }
        __syncthreads();
    }
    if (tid == 0) out[0] = ss[0] / fmaxf(sc[0], 1.f);
}

// ---------------------------------------------------------------------------
// Workspace layout (requires ws_size >= ~16.9 MB):
//   [0)            8,388,608 B : Wt   bf16  (W_enc transposed)
//   [8,388,608)    8,388,608 B : Abf  bf16  (context)
//   [16,777,216)      32,768 B : targets (int32 x 8192)
//   [16,809,984)       2,048 B : partials (float x 512)
// ---------------------------------------------------------------------------
extern "C" void kernel_launch(void* const* d_in, const int* in_sizes, int n_in,
                              void* d_out, int out_size, void* d_ws, size_t ws_size,
                              hipStream_t stream) {
    (void)in_sizes; (void)n_in; (void)out_size; (void)ws_size;
    const float* feats    = (const float*)d_in[0];
    const float* context  = (const float*)d_in[1];
    const int*   lens     = (const int*)  d_in[2];
    const float* proj     = (const float*)d_in[3];
    const float* codebook = (const float*)d_in[4];
    const float* W_enc    = (const float*)d_in[5];
    const float* b_enc    = (const float*)d_in[6];

    char* ws = (char*)d_ws;
    u16*   Wt       = (u16*)ws;
    u16*   Abf      = (u16*)(ws + 8388608);
    int*   targets  = (int*)(ws + 16777216);
    float* partials = (float*)(ws + 16777216 + 32768);

    cvtA_kernel<<<4096, 256, 0, stream>>>(context, Abf);
    dim3 gT(256, 16);
    cvtBT_kernel<<<gT, 256, 0, stream>>>(W_enc, Wt);
    targets_kernel<<<512, 256, 0, stream>>>(feats, lens, proj, codebook, targets);
    gemmce_kernel<<<256, 256, 0, stream>>>(Abf, Wt, b_enc, targets, partials);
    finalize_kernel<<<1, 256, 0, stream>>>(partials, (float*)d_out);
}
